// RNN_2903397893063
// MI455X (gfx1250) — matrix operand order repaired and checked by reading, not compiled, not run on hardware
//
#include <hip/hip_runtime.h>
#include <hip/hip_bf16.h>

// Tanh-RNN on MI455X (gfx1250, wave32, WMMA).
//   K1 (parallel):  preT[t] = (Wxh @ x_t + bh)^T      -> ws (f32, [S,B,H])
//   K2 (serial):    h = tanh(pre[t] + Whh @ h); hsT[t]=h^T (bf16 -> ws), h_last -> out
//                   single persistent workgroup, Whh in VGPRs, double-buffered h in LDS
//   K3 (parallel):  y[t] = Why @ hs[t] + by           -> out (f32)

#define S_LEN   2048
#define IN_DIM  256
#define HID     512
#define OUT_DIM 256
#define BATCH   64

#define LDX 264   // LDS row stride (halves) for x_t staging: 528B -> 4-bank skew
#define LDH 520   // LDS row stride (halves) for h staging:  1040B -> 4-bank skew

typedef __attribute__((ext_vector_type(16))) __bf16 v16bf;
typedef __attribute__((ext_vector_type(8)))  __bf16 v8bf;
typedef __attribute__((ext_vector_type(8)))  float  v8f;

// Single-instruction transcendental tanh (gfx1250 V_TANH_F32).
// Trailing v_nop satisfies the TRANS-op result-use hazard (1 independent op).
__device__ __forceinline__ float fast_tanh(float x) {
    float r;
    asm("v_tanh_f32 %0, %1\n\tv_nop" : "=v"(r) : "v"(x));
    return r;
}

// ---- A tile (16x32 bf16) from row-major f32 matrix, contiguous k = 16*(lane/16) + i ----
__device__ __forceinline__ v16bf load_a_tile_f32(const float* __restrict__ M, int ld,
                                                 int mrow, int kbase, int lane) {
    const int m  = lane & 15;
    const int kb = (lane < 16) ? 0 : 16;
    const float* row = M + (size_t)(mrow + m) * ld + kbase;
    v16bf a;
#pragma unroll
    for (int i = 0; i < 16; ++i) {
        const int k = kb + i;
        a[i] = (__bf16)row[k];
    }
    return a;
}

// ---- B tile (32x16 bf16) from LDS holding the operand transposed: hT[n][k] ----
__device__ __forceinline__ v16bf load_b_tile_lds(const __bf16* hT, int ldw,
                                                 int kbase, int nbase, int lane) {
    const int n  = lane & 15;
    const int k0 = (lane < 16) ? 0 : 16;
    const v8bf* p = (const v8bf*)(hT + (size_t)(nbase + n) * ldw + kbase + k0);
    const v8bf lo = p[0];
    const v8bf hi = p[1];
    v16bf b;
#pragma unroll
    for (int i = 0; i < 8; ++i) { b[i] = lo[i]; b[i + 8] = hi[i]; }
    return b;
}

#define WMMA_BF16(A, B, C) \
    __builtin_amdgcn_wmma_f32_16x16x32_bf16(false, (A), false, (B), (short)0, (C), false, false)

// =====================  K1: preT[t] = (Wxh @ x_t + bh)^T  =====================
__global__ __launch_bounds__(1024) void rnn_pre_kernel(const float* __restrict__ x,
                                                       const float* __restrict__ Wxh,
                                                       const float* __restrict__ bh,
                                                       float* __restrict__ preT) {
    __shared__ __bf16 xT[BATCH * LDX];
    const int t = blockIdx.x;
    const int tid = threadIdx.x, wave = tid >> 5, lane = tid & 31;

    const float* xt = x + (size_t)t * IN_DIM * BATCH;
    for (int idx = tid; idx < IN_DIM * BATCH; idx += 1024) {
        const int k = idx >> 6, n = idx & 63;
        xT[n * LDX + k] = (__bf16)xt[idx];
    }

    const int mrow  = wave * 16;
    const int m_off = (lane < 16) ? 0 : 8;
    const int n     = lane & 15;

    v16bf a[8];
#pragma unroll
    for (int kt = 0; kt < 8; ++kt) a[kt] = load_a_tile_f32(Wxh, IN_DIM, mrow, kt * 32, lane);

    float bv[8];
#pragma unroll
    for (int r = 0; r < 8; ++r) bv[r] = bh[mrow + r + m_off];

    v8f acc[4] = {};
    __syncthreads();
#pragma unroll
    for (int kt = 0; kt < 8; ++kt) {
        const v16bf b0 = load_b_tile_lds(xT, LDX, kt * 32,  0, lane);
        const v16bf b1 = load_b_tile_lds(xT, LDX, kt * 32, 16, lane);
        const v16bf b2 = load_b_tile_lds(xT, LDX, kt * 32, 32, lane);
        const v16bf b3 = load_b_tile_lds(xT, LDX, kt * 32, 48, lane);
        acc[0] = WMMA_BF16(a[kt], b0, acc[0]);
        acc[1] = WMMA_BF16(a[kt], b1, acc[1]);
        acc[2] = WMMA_BF16(a[kt], b2, acc[2]);
        acc[3] = WMMA_BF16(a[kt], b3, acc[3]);
    }

    // preT layout [B, H]: a lane's 8 M-values are contiguous -> vector stores
#pragma unroll
    for (int nb = 0; nb < 4; ++nb) {
        v8f v;
#pragma unroll
        for (int r = 0; r < 8; ++r) v[r] = acc[nb][r] + bv[r];
        *(v8f*)(preT + ((size_t)t * BATCH + nb * 16 + n) * HID + mrow + m_off) = v;
    }
}

// =====================  K2: serial recurrence  =====================
__global__ __launch_bounds__(1024) void rnn_rec_kernel(const float* __restrict__ h0,
                                                       const float* __restrict__ Whh,
                                                       const float* __restrict__ preT,
                                                       __bf16* __restrict__ hsT,
                                                       float* __restrict__ h_last) {
    __shared__ __bf16 hT[2][BATCH * LDH];    // double-buffered h (2 x 66.6 KB of 320 KB)
    const int tid = threadIdx.x, wave = tid >> 5, lane = tid & 31;
    const int mrow  = wave * 16;
    const int m_off = (lane < 16) ? 0 : 8;
    const int n     = lane & 15;

    // Pin Whh row-block in registers for the whole sequence (16 A-tiles = 128 VGPRs).
    v16bf a[16];
#pragma unroll
    for (int kt = 0; kt < 16; ++kt) a[kt] = load_a_tile_f32(Whh, HID, mrow, kt * 32, lane);

    for (int idx = tid; idx < HID * BATCH; idx += 1024) {
        const int k = idx >> 6, nn = idx & 63;
        hT[0][nn * LDH + k] = (__bf16)h0[idx];
    }
    __syncthreads();

    for (int t = 0; t < S_LEN; ++t) {
        const __bf16* rd = hT[t & 1];
        __bf16*       wr = hT[(t + 1) & 1];
        const float* prt = preT + (size_t)t * BATCH * HID;
        if (t + 1 < S_LEN)
            __builtin_prefetch(prt + (size_t)BATCH * HID, 0, 1);   // global_prefetch_b8

        // Issue pre[t] loads now (8 x b128); consume after kt=2 (latency hidden).
        v8f pr[4];
#pragma unroll
        for (int nb = 0; nb < 4; ++nb)
            pr[nb] = *(const v8f*)(prt + (size_t)(nb * 16 + n) * HID + mrow + m_off);

        v8f acc[4] = {};
#pragma unroll
        for (int kt = 0; kt < 3; ++kt) {
            const v16bf b0 = load_b_tile_lds(rd, LDH, kt * 32,  0, lane);
            const v16bf b1 = load_b_tile_lds(rd, LDH, kt * 32, 16, lane);
            const v16bf b2 = load_b_tile_lds(rd, LDH, kt * 32, 32, lane);
            const v16bf b3 = load_b_tile_lds(rd, LDH, kt * 32, 48, lane);
            acc[0] = WMMA_BF16(a[kt], b0, acc[0]);
            acc[1] = WMMA_BF16(a[kt], b1, acc[1]);
            acc[2] = WMMA_BF16(a[kt], b2, acc[2]);
            acc[3] = WMMA_BF16(a[kt], b3, acc[3]);
        }
        // Fold in pre[t] here: ends pr's live range early (frees 32 VGPRs, no spills).
#pragma unroll
        for (int nb = 0; nb < 4; ++nb) acc[nb] = acc[nb] + pr[nb];
#pragma unroll
        for (int kt = 3; kt < 16; ++kt) {
            const v16bf b0 = load_b_tile_lds(rd, LDH, kt * 32,  0, lane);
            const v16bf b1 = load_b_tile_lds(rd, LDH, kt * 32, 16, lane);
            const v16bf b2 = load_b_tile_lds(rd, LDH, kt * 32, 32, lane);
            const v16bf b3 = load_b_tile_lds(rd, LDH, kt * 32, 48, lane);
            acc[0] = WMMA_BF16(a[kt], b0, acc[0]);
            acc[1] = WMMA_BF16(a[kt], b1, acc[1]);
            acc[2] = WMMA_BF16(a[kt], b2, acc[2]);
            acc[3] = WMMA_BF16(a[kt], b3, acc[3]);
        }

        // tanh, then vector stores: LDS (next buffer) + global hsT, 4 x b128 each.
        __bf16* hst = hsT + (size_t)t * BATCH * HID;
#pragma unroll
        for (int nb = 0; nb < 4; ++nb) {
            const int nc = nb * 16 + n;
            v8bf hv;
#pragma unroll
            for (int r = 0; r < 8; ++r) hv[r] = (__bf16)fast_tanh(acc[nb][r]);
            *(v8bf*)&wr[nc * LDH + mrow + m_off] = hv;                       // h -> LDS
            *(v8bf*)(hst + (size_t)nc * HID + mrow + m_off) = hv;            // h -> global
        }
        if (t == S_LEN - 1) {
#pragma unroll
            for (int nb = 0; nb < 4; ++nb)
#pragma unroll
                for (int r = 0; r < 8; ++r)
                    h_last[(size_t)(mrow + r + m_off) * BATCH + nb * 16 + n] =
                        fast_tanh(acc[nb][r]);
        }
        __syncthreads();   // single barrier: writes visible + old-buf reads retired
    }
}

// =====================  K3: y[t] = Why @ hs[t] + by  =====================
__global__ __launch_bounds__(512) void rnn_out_kernel(const __bf16* __restrict__ hsT,
                                                      const float* __restrict__ Why,
                                                      const float* __restrict__ by,
                                                      float* __restrict__ y) {
    __shared__ __bf16 hT[BATCH * LDH];
    const int t = blockIdx.x;
    const int tid = threadIdx.x, wave = tid >> 5, lane = tid & 31;
    const int orow  = wave * 16;
    const int m_off = (lane < 16) ? 0 : 8;
    const int n     = lane & 15;

    // hsT is already [B, H]: stage into LDS with 16-byte block copies.
    const v8bf* src = (const v8bf*)(hsT + (size_t)t * BATCH * HID);
    for (int c = tid; c < (BATCH * HID) / 8; c += 512) {
        const int nn = c >> 6;            // 64 chunks of 8 halves per row of 512
        const int k  = (c & 63) * 8;
        *(v8bf*)&hT[nn * LDH + k] = src[c];
    }

    v16bf a[16];
#pragma unroll
    for (int kt = 0; kt < 16; ++kt) a[kt] = load_a_tile_f32(Why, HID, orow, kt * 32, lane);

    float bv[8];
#pragma unroll
    for (int r = 0; r < 8; ++r) bv[r] = by[orow + r + m_off];

    v8f acc[4] = {};
    __syncthreads();
#pragma unroll
    for (int kt = 0; kt < 16; ++kt) {
        const v16bf b0 = load_b_tile_lds(hT, LDH, kt * 32,  0, lane);
        const v16bf b1 = load_b_tile_lds(hT, LDH, kt * 32, 16, lane);
        const v16bf b2 = load_b_tile_lds(hT, LDH, kt * 32, 32, lane);
        const v16bf b3 = load_b_tile_lds(hT, LDH, kt * 32, 48, lane);
        acc[0] = WMMA_BF16(a[kt], b0, acc[0]);
        acc[1] = WMMA_BF16(a[kt], b1, acc[1]);
        acc[2] = WMMA_BF16(a[kt], b2, acc[2]);
        acc[3] = WMMA_BF16(a[kt], b3, acc[3]);
    }

    float* yt = y + ((size_t)t * OUT_DIM + orow) * BATCH;
#pragma unroll
    for (int nb = 0; nb < 4; ++nb)
#pragma unroll
        for (int r = 0; r < 8; ++r)
            yt[(size_t)(r + m_off) * BATCH + nb * 16 + n] = acc[nb][r] + bv[r];
}

extern "C" void kernel_launch(void* const* d_in, const int* in_sizes, int n_in,
                              void* d_out, int out_size, void* d_ws, size_t ws_size,
                              hipStream_t stream) {
    (void)in_sizes; (void)n_in; (void)out_size; (void)ws_size;
    const float* x   = (const float*)d_in[0];   // [S, I, B]
    const float* h0  = (const float*)d_in[1];   // [H, B]
    const float* Wxh = (const float*)d_in[2];   // [H, I]
    const float* Whh = (const float*)d_in[3];   // [H, H]
    const float* bh  = (const float*)d_in[4];   // [H, 1]
    const float* Why = (const float*)d_in[5];   // [O, H]
    const float* by  = (const float*)d_in[6];   // [O, 1]

    float*  y      = (float*)d_out;                                  // [S, O, B]
    float*  h_last = y + (size_t)S_LEN * OUT_DIM * BATCH;            // [H, B]

    float*  preT = (float*)d_ws;                                     // [S, B, H] f32
    __bf16* hsT  = (__bf16*)((char*)d_ws +
                   (size_t)S_LEN * HID * BATCH * sizeof(float));     // [S, B, H] bf16

    rnn_pre_kernel<<<dim3(S_LEN), dim3(1024), 0, stream>>>(x, Wxh, bh, preT);
    rnn_rec_kernel<<<dim3(1),     dim3(1024), 0, stream>>>(h0, Whh, preT, hsT, h_last);
    rnn_out_kernel<<<dim3(S_LEN), dim3(512),  0, stream>>>(hsT, Why, by, y);
}
